// Mamba_68642167324761
// MI455X (gfx1250) — matrix operand order repaired and checked by reading, not compiled, not run on hardware
//
#include <hip/hip_runtime.h>
#include <hip/hip_bf16.h>
#include <stdint.h>

#define D_MODEL 768
#define D_INNER 1536
#define D_STATE 16
#define DT_RANK 48
#define D_CONV  4
#define BB      2
#define LL      1024
#define NL      2
#define ROWS    (BB*LL)                  // 2048
#define DBC_W   (DT_RANK + 2*D_STATE)    // 80
#define DBC_PAD 128                      // padded N for x_proj GEMM

typedef __attribute__((ext_vector_type(8)))  unsigned short u16x8;
typedef __attribute__((ext_vector_type(16))) unsigned short u16x16;
typedef __attribute__((ext_vector_type(16))) __bf16         bf16x16;
typedef __attribute__((ext_vector_type(8)))  float          f32x8;

__device__ __forceinline__ unsigned short f32_to_bf16(float f) {
  unsigned int u = __builtin_bit_cast(unsigned int, f);
  u += 0x7FFFu + ((u >> 16) & 1u);          // round-to-nearest-even
  return (unsigned short)(u >> 16);
}

// CDNA5 async global->LDS copy (16B per lane), tracked by ASYNCcnt.
__device__ __forceinline__ void async_b128_to_lds(unsigned lds_off, const void* gaddr) {
  asm volatile("global_load_async_to_lds_b128 %0, %1, off"
               :: "v"(lds_off), "v"(gaddr) : "memory");
}
__device__ __forceinline__ void wait_asynccnt0() {
  asm volatile("s_wait_asynccnt 0x0" ::: "memory");
}

// ------------------------------------------------ f32 [K,N] -> bf16 [Npad,K] (transposed, padded)
__global__ __launch_bounds__(256)
void cvt_transpose_bf16(const float* __restrict__ W, unsigned short* __restrict__ WT,
                        int K, int N, int Npad) {
  int idx = blockIdx.x * 256 + threadIdx.x;       // over Npad*K, writes contiguous
  if (idx >= Npad * K) return;
  int n = idx / K;
  int k = idx - n * K;
  float v = (n < N) ? W[(size_t)k * N + n] : 0.0f;
  WT[(size_t)n * K + k] = f32_to_bf16(v);
}

// ------------------------------------------------ RMSNorm -> bf16
__global__ __launch_bounds__(256)
void rmsnorm_bf16(const float* __restrict__ x, const float* __restrict__ w,
                  unsigned short* __restrict__ out) {
  __shared__ float red[256];
  int row = blockIdx.x;
  const float* xr = x + (size_t)row * D_MODEL;
  float ss = 0.f;
  for (int i = threadIdx.x; i < D_MODEL; i += 256) { float v = xr[i]; ss += v * v; }
  red[threadIdx.x] = ss;
  __syncthreads();
  for (int off = 128; off > 0; off >>= 1) {
    if ((int)threadIdx.x < off) red[threadIdx.x] += red[threadIdx.x + off];
    __syncthreads();
  }
  float rstd = rsqrtf(red[0] / (float)D_MODEL + 1e-5f);
  for (int i = threadIdx.x; i < D_MODEL; i += 256)
    out[(size_t)row * D_MODEL + i] = f32_to_bf16(xr[i] * rstd * w[i]);
}

// ------------------------------------------------ bf16 WMMA GEMM
// C[M,N] = A[M,K] (bf16 row-major) x W (given pre-TRANSPOSED: WT[Npad,K] bf16).
// Block: 256 thr (8 waves), tile 128(M) x 64(N), K-step 32. Guard-free main loop
// (M mult of 128, Npad mult of 64, K mult of 32); epilogue guards col < N.
// EPI: 0 = plain store, 1 = C = acc + R (R may alias C)
template<int EPI>
__global__ __launch_bounds__(256)
void gemm_bf16(const unsigned short* __restrict__ A,
               const unsigned short* __restrict__ WT,
               float* C, const float* R,
               int M_, int N_, int K_) {
  __shared__ unsigned short lds_w[64 * 40];       // 64 cols x 32 K (+pad: 40)
  const int tid  = threadIdx.x;
  const int lane = tid & 31;
  const int wave = tid >> 5;
  const int mBase = blockIdx.y * 128;
  const int nBase = blockIdx.x * 64;
  const int halfsel = lane >> 4;
  const int l16     = lane & 15;
  const int ka      = halfsel ? 16 : 0;   // A frag: K 0..15 vs 16..31
  const int koffB   = halfsel ? 16 : 0;   // B frag: K 0..15 vs 16..31

  // A fragment source row (direct from global, no reuse within block)
  const unsigned short* aRow = A + (size_t)(mBase + wave * 16 + l16) * K_;
  // W staging: this thread async-copies WT[nBase + tid/4][ (tid%4)*8 .. +8 ) each K-step
  const int sn = tid >> 2;                // 0..63
  const int sc = (tid & 3) * 8;           // 0,8,16,24
  const unsigned short* wRow = WT + (size_t)(nBase + sn) * K_ + sc;
  const unsigned ldsDst = (unsigned)(size_t)(&lds_w[sn * 40 + sc]);

  f32x8 acc[4];
  #pragma unroll
  for (int i = 0; i < 4; ++i)
    #pragma unroll
    for (int j = 0; j < 8; ++j) acc[i][j] = 0.0f;

  for (int k0 = 0; k0 < K_; k0 += 32) {
    __syncthreads();                       // prev-iter B reads done before overwrite
    async_b128_to_lds(ldsDst, (const void*)(wRow + k0));

    // A fragment: two b128 global loads (overlap with async copy)
    u16x16 araw;
    {
      u16x8 lo = *(const u16x8*)(aRow + k0 + ka);
      u16x8 hi = *(const u16x8*)(aRow + k0 + ka + 8);
      #pragma unroll
      for (int j = 0; j < 8; ++j) { araw[j] = lo[j]; araw[j + 8] = hi[j]; }
    }
    bf16x16 afrag = __builtin_bit_cast(bf16x16, araw);
    if (k0 + 32 < K_) __builtin_prefetch(aRow + k0 + 32, 0, 1);

    wait_asynccnt0();                      // my wave's LDS writes landed
    __syncthreads();                       // everyone's landed

    #pragma unroll
    for (int nt = 0; nt < 4; ++nt) {
      int col = nt * 16 + l16;
      u16x16 braw;
      u16x8 lo = *(const u16x8*)(&lds_w[col * 40 + koffB]);
      u16x8 hi = *(const u16x8*)(&lds_w[col * 40 + koffB + 8]);
      #pragma unroll
      for (int j = 0; j < 8; ++j) { braw[j] = lo[j]; braw[j + 8] = hi[j]; }
      bf16x16 bfrag = __builtin_bit_cast(bf16x16, braw);
      acc[nt] = __builtin_amdgcn_wmma_f32_16x16x32_bf16(
          false, afrag, false, bfrag, (short)0, acc[nt], false, false);
    }
  }

  // epilogue: D layout: VGPR v -> row (halfsel*8 + v), col = lane&15
  const int rowB = mBase + wave * 16 + halfsel * 8;
  #pragma unroll
  for (int nt = 0; nt < 4; ++nt) {
    int col = nBase + nt * 16 + l16;
    if (col < N_) {
      #pragma unroll
      for (int v = 0; v < 8; ++v) {
        size_t idx = (size_t)(rowB + v) * N_ + col;
        float r = acc[nt][v];
        if (EPI == 1) r += R[idx];
        C[idx] = r;
      }
    }
  }
}

// ------------------------------------------------ causal depthwise conv + SiLU
__global__ __launch_bounds__(256)
void conv_silu(const float* __restrict__ xz, const float* __restrict__ cw,
               const float* __restrict__ cb, float* __restrict__ uf,
               unsigned short* __restrict__ ub) {
  int idx = blockIdx.x * 256 + threadIdx.x;   // over ROWS*D_INNER
  int d  = idx % D_INNER;
  int bl = idx / D_INNER;
  int l  = bl % LL;
  int b  = bl / LL;
  float a = cb[d];
  #pragma unroll
  for (int j = 0; j < D_CONV; ++j) {
    int t = l - (D_CONV - 1) + j;
    if (t >= 0)
      a += xz[(size_t)(b * LL + t) * (2 * D_INNER) + d] * cw[d * D_CONV + j];
  }
  float s = a / (1.0f + expf(-a));            // SiLU
  uf[idx] = s;
  ub[idx] = f32_to_bf16(s);
}

// ------------------------------------------------ dt projection + softplus
__global__ __launch_bounds__(256)
void dt_softplus(const float* __restrict__ dbc, const float* __restrict__ dtw,
                 const float* __restrict__ dtb, float* __restrict__ dlt) {
  int row = blockIdx.x;
  int d   = blockIdx.y * 256 + threadIdx.x;
  float a = dtb[d];
  const float* r = dbc + (size_t)row * DBC_W;
  #pragma unroll 8
  for (int j = 0; j < DT_RANK; ++j) a += r[j] * dtw[(size_t)j * D_INNER + d];
  float sp = (a > 20.0f) ? a : log1pf(expf(a));
  dlt[(size_t)row * D_INNER + d] = sp;
}

// ------------------------------------------------ selective scan (register-resident state)
__global__ __launch_bounds__(256)
void ssm_scan(const float* __restrict__ dlt, const float* __restrict__ uf,
              const float* __restrict__ dbc, const float* __restrict__ xz,
              const float* __restrict__ A_log, const float* __restrict__ Dp,
              unsigned short* __restrict__ yb) {
  __shared__ float bc[2 * D_STATE];
  const int b = blockIdx.x / (D_INNER / 256);
  const int d = (blockIdx.x % (D_INNER / 256)) * 256 + threadIdx.x;
  float Ac[D_STATE];
  #pragma unroll
  for (int s = 0; s < D_STATE; ++s) Ac[s] = -expf(A_log[(size_t)d * D_STATE + s]);
  const float Dd = Dp[d];
  float h[D_STATE];
  #pragma unroll
  for (int s = 0; s < D_STATE; ++s) h[s] = 0.0f;

  for (int t = 0; t < LL; ++t) {
    size_t row = (size_t)b * LL + t;
    if ((int)threadIdx.x < 2 * D_STATE)
      bc[threadIdx.x] = dbc[row * DBC_W + DT_RANK + threadIdx.x];
    __syncthreads();
    float dtv = dlt[row * D_INNER + d];
    float ut  = uf [row * D_INNER + d];
    float y = 0.0f;
    #pragma unroll
    for (int s = 0; s < D_STATE; ++s) {
      float dA = expf(dtv * Ac[s]);
      h[s] = dA * h[s] + (dtv * bc[s]) * ut;
      y += h[s] * bc[D_STATE + s];
    }
    y += Dd * ut;
    float z = xz[row * (2 * D_INNER) + D_INNER + d];
    float g = z / (1.0f + expf(-z));           // SiLU(z)
    yb[row * D_INNER + d] = f32_to_bf16(y * g);
    __syncthreads();
  }
}

// ------------------------------------------------ host
extern "C" void kernel_launch(void* const* d_in, const int* in_sizes, int n_in,
                              void* d_out, int out_size, void* d_ws, size_t ws_size,
                              hipStream_t stream) {
  const float* x      = (const float*)d_in[0];
  const float* norm_w = (const float*)d_in[1];
  const float* in_w   = (const float*)d_in[2];
  const float* conv_w = (const float*)d_in[3];
  const float* conv_b = (const float*)d_in[4];
  const float* xproj_w= (const float*)d_in[5];
  const float* dt_w   = (const float*)d_in[6];
  const float* dt_b   = (const float*)d_in[7];
  const float* A_log  = (const float*)d_in[8];
  const float* D_par  = (const float*)d_in[9];
  const float* out_w  = (const float*)d_in[10];
  float* xcur = (float*)d_out;               // running residual buffer

  char* ws = (char*)d_ws;
  size_t off = 0;
  auto alloc = [&](size_t bytes) -> void* {
    void* p = ws + off;
    off = (off + bytes + 255) & ~(size_t)255;
    return p;
  };
  unsigned short* in_wT  = (unsigned short*)alloc((size_t)NL*2*D_INNER*D_MODEL*2); // [3072,768]
  unsigned short* xp_wT  = (unsigned short*)alloc((size_t)NL*DBC_PAD*D_INNER*2);   // [128,1536]
  unsigned short* out_wT = (unsigned short*)alloc((size_t)NL*D_MODEL*D_INNER*2);   // [768,1536]
  unsigned short* h_b    = (unsigned short*)alloc((size_t)ROWS*D_MODEL*2);
  float*          xz     = (float*)         alloc((size_t)ROWS*2*D_INNER*4);
  float*          uf     = (float*)         alloc((size_t)ROWS*D_INNER*4);
  unsigned short* ub     = (unsigned short*)alloc((size_t)ROWS*D_INNER*2);
  float*          dbc    = (float*)         alloc((size_t)ROWS*DBC_W*4);
  float*          dlt    = (float*)         alloc((size_t)ROWS*D_INNER*4);
  unsigned short* yb     = (unsigned short*)alloc((size_t)ROWS*D_INNER*2);

  hipMemcpyAsync(xcur, x, (size_t)ROWS*D_MODEL*sizeof(float),
                 hipMemcpyDeviceToDevice, stream);

  // one-time (per call) weight convert + transpose + pad
  for (int l = 0; l < NL; ++l) {
    int n1 = 2*D_INNER*D_MODEL;   // 3072*768
    cvt_transpose_bf16<<<(n1+255)/256, 256, 0, stream>>>(
        in_w + (size_t)l*D_MODEL*2*D_INNER, in_wT + (size_t)l*n1,
        D_MODEL, 2*D_INNER, 2*D_INNER);
    int n2 = DBC_PAD*D_INNER;     // 128*1536
    cvt_transpose_bf16<<<(n2+255)/256, 256, 0, stream>>>(
        xproj_w + (size_t)l*D_INNER*DBC_W, xp_wT + (size_t)l*n2,
        D_INNER, DBC_W, DBC_PAD);
    int n3 = D_MODEL*D_INNER;     // 768*1536
    cvt_transpose_bf16<<<(n3+255)/256, 256, 0, stream>>>(
        out_w + (size_t)l*D_INNER*D_MODEL, out_wT + (size_t)l*n3,
        D_INNER, D_MODEL, D_MODEL);
  }

  for (int l = 0; l < NL; ++l) {
    rmsnorm_bf16<<<ROWS, 256, 0, stream>>>(xcur, norm_w + (size_t)l*D_MODEL, h_b);

    gemm_bf16<0><<<dim3((2*D_INNER)/64, ROWS/128), 256, 0, stream>>>(
        h_b, in_wT + (size_t)l*2*D_INNER*D_MODEL, xz, nullptr,
        ROWS, 2*D_INNER, D_MODEL);

    conv_silu<<<(ROWS*D_INNER)/256, 256, 0, stream>>>(
        xz, conv_w + (size_t)l*D_INNER*D_CONV, conv_b + (size_t)l*D_INNER, uf, ub);

    gemm_bf16<0><<<dim3(DBC_PAD/64, ROWS/128), 256, 0, stream>>>(
        ub, xp_wT + (size_t)l*DBC_PAD*D_INNER, dbc, nullptr,
        ROWS, DBC_W, D_INNER);

    dt_softplus<<<dim3(ROWS, D_INNER/256), 256, 0, stream>>>(
        dbc, dt_w + (size_t)l*DT_RANK*D_INNER, dt_b + (size_t)l*D_INNER, dlt);

    ssm_scan<<<BB*(D_INNER/256), 256, 0, stream>>>(
        dlt, uf, dbc, xz, A_log + (size_t)l*D_INNER*D_STATE,
        D_par + (size_t)l*D_INNER, yb);

    gemm_bf16<1><<<dim3(D_MODEL/64, ROWS/128), 256, 0, stream>>>(
        yb, out_wT + (size_t)l*D_MODEL*D_INNER, xcur, xcur,
        ROWS, D_MODEL, D_INNER);
  }
}
